// Model_34316788695805
// MI455X (gfx1250) — matrix operand order repaired and checked by reading, not compiled, not run on hardware
//
#include <hip/hip_runtime.h>

typedef __attribute__((ext_vector_type(8)))  _Float16 v8h;
typedef __attribute__((ext_vector_type(16))) _Float16 v16h;
typedef __attribute__((ext_vector_type(8)))  float    v8f;

#define T_ 256
#define F_ 32
#define C_ 8
#define V_ 10000
#define VPAD_ 10048   /* V rounded up to multiple of 64 for unguarded 16x64 tiles */
#define H_ 1024
#define E_ 256
#define EOS_ 2

// ---------------- f32 -> f16 conversion ----------------
__global__ void k_f32_to_f16(const float* __restrict__ s, _Float16* __restrict__ d, int n) {
    int i = blockIdx.x * 256 + threadIdx.x;
    if (i < n) d[i] = (_Float16)s[i];
}

// ---------------- h0 = mean over C of token_ctx[xtoken_ids] ----------------
__global__ void k_h0(const float* __restrict__ tctx, const int* __restrict__ sent,
                     float* __restrict__ h, _Float16* __restrict__ h16) {
    int t = blockIdx.x;
    int j = blockIdx.y * 256 + threadIdx.x;
    float acc = 0.f;
#pragma unroll
    for (int c = 0; c < C_; ++c) {
        int xid = sent[(t * C_ + c) * 3 + 1];
        acc += tctx[(size_t)xid * H_ + j];
    }
    acc *= (1.f / C_);
    h[(size_t)t * H_ + j]   = acc;
    h16[(size_t)t * H_ + j] = (_Float16)acc;
}

// ---------------- decoder init: c0 = out_chars[0] broadcast ----------------
__global__ void k_c0(const int* __restrict__ out_chars, int* __restrict__ c,
                     int* __restrict__ done) {
    int t = threadIdx.x;
    int c0 = out_chars[0];
    c[t] = c0;
    done[t] = (c0 == EOS_) ? 1 : 0;
}

// ---------------- embedding gathers (f16) ----------------
__global__ void k_embed_enc(const int* __restrict__ sent, const _Float16* __restrict__ emb,
                            _Float16* __restrict__ x, int step) {
    int t = blockIdx.x, e = threadIdx.x;
    int ch = sent[(t * C_ + step) * 3 + 2];
    x[(size_t)t * E_ + e] = emb[(size_t)ch * E_ + e];
}

__global__ void k_embed_dec(const int* __restrict__ c, const _Float16* __restrict__ emb,
                            _Float16* __restrict__ x) {
    int t = blockIdx.x, e = threadIdx.x;
    int ch = c[t];
    x[(size_t)t * E_ + e] = emb[(size_t)ch * E_ + e];
}

// ---------------- fragment helpers ----------------
__device__ __forceinline__ v16h frag16(v8h lo, v8h hi) {
    v16h r;
#pragma unroll
    for (int q = 0; q < 8; ++q) { r[q] = lo[q]; r[q + 8] = hi[q]; }
    return r;
}

__device__ __forceinline__ void epilogue(v8f acc, int col, int mbase, int lg,
                                         const float* __restrict__ bias,
                                         const int* __restrict__ done,
                                         const float* __restrict__ prev,
                                         float* __restrict__ out,
                                         unsigned long long ldo, int nt) {
    float bv = bias ? bias[col] : 0.f;
#pragma unroll
    for (int j = 0; j < 8; ++j) {
        int r = mbase + lg * 8 + j;
        float v = acc[j] + bv;
        if (done && done[r]) v = prev ? prev[(size_t)r * ldo + col] : 0.f;
        float* dst = out + (size_t)r * ldo + col;
        if (nt) __builtin_nontemporal_store(v, dst);
        else    *dst = v;
    }
}

// ---------------- WMMA GEMM: out[M,ldo] = A[M,K] * W[N,K]^T (+bias, +done/prev mask) --------
// One wave = one 16x64 output tile (4 accumulators): each A fragment feeds 4 WMMAs, so
// loads/wmma = 2.5 b128 instead of 4 and A request traffic drops 4x. The k-loop is fully
// unconditional (W is padded to a multiple of 64 rows by the host, so B loads for edge
// subtiles stay in-bounds); only the epilogue carries the wave-uniform, unrolled
// "subtile < Nreal" store guard. EXEC is all-ones for every executed WMMA.
// A fragment (16-bit A 16x32): lane m = lane&15; lanes0-15 hold K 0-15,
// lanes16-31 hold K 16-31 (two 16B loads/lane).
// B fragment (32x16 = W^T tile): lane col n = lane&15; lanes0-15 hold K 0-15,
// lanes16-31 hold K 16-31 -> one contiguous 32B load from W row n.
__global__ void __launch_bounds__(32)
k_wmma_gemm(const _Float16* __restrict__ A,
            const _Float16* __restrict__ W,       // Npad x K, Npad % 64 == 0
            const float* __restrict__ bias,       // len >= Nreal, nullable
            const int* __restrict__ done,         // len M, nullable
            const float* __restrict__ prev,       // M x ldo, nullable
            float* __restrict__ out,              // M x ldo
            int K, int Nreal, unsigned long long ldo, int nt_store) {
    int lane = threadIdx.x;
    int lh = lane & 15;     // row-in-tile (A) / col-in-tile (B)
    int lg = lane >> 4;     // half-wave id
    int mbase  = blockIdx.y * 16;
    int nbase0 = blockIdx.x * 64;

    v8f acc0 = {}, acc1 = {}, acc2 = {}, acc3 = {};
    const _Float16* arow = A + (size_t)(mbase + lh) * K;
    const _Float16* br0  = W + (size_t)(nbase0 + lh) * K;
    const _Float16* br1  = br0 + (size_t)16 * K;
    const _Float16* br2  = br0 + (size_t)32 * K;
    const _Float16* br3  = br0 + (size_t)48 * K;

    int kiters = K >> 5;
    for (int kt = 0; kt < kiters; ++kt) {
        int ko = kt * 32;
        v16h a  = frag16(*(const v8h*)(arow + ko + lg * 16),
                         *(const v8h*)(arow + ko + lg * 16 + 8));
        v16h b0 = frag16(*(const v8h*)(br0 + ko + lg * 16),
                         *(const v8h*)(br0 + ko + lg * 16 + 8));
        v16h b1 = frag16(*(const v8h*)(br1 + ko + lg * 16),
                         *(const v8h*)(br1 + ko + lg * 16 + 8));
        v16h b2 = frag16(*(const v8h*)(br2 + ko + lg * 16),
                         *(const v8h*)(br2 + ko + lg * 16 + 8));
        v16h b3 = frag16(*(const v8h*)(br3 + ko + lg * 16),
                         *(const v8h*)(br3 + ko + lg * 16 + 8));
        acc0 = __builtin_amdgcn_wmma_f32_16x16x32_f16(false, a, false, b0, (short)0, acc0, false, false);
        acc1 = __builtin_amdgcn_wmma_f32_16x16x32_f16(false, a, false, b1, (short)0, acc1, false, false);
        acc2 = __builtin_amdgcn_wmma_f32_16x16x32_f16(false, a, false, b2, (short)0, acc2, false, false);
        acc3 = __builtin_amdgcn_wmma_f32_16x16x32_f16(false, a, false, b3, (short)0, acc3, false, false);
    }

    // C/D layout: VGPR j -> M = j (lanes 0-15) or j+8 (lanes 16-31); N = lane&15.
    if (nbase0 +  0 < Nreal) epilogue(acc0, nbase0 +  0 + lh, mbase, lg, bias, done, prev, out, ldo, nt_store);
    if (nbase0 + 16 < Nreal) epilogue(acc1, nbase0 + 16 + lh, mbase, lg, bias, done, prev, out, ldo, nt_store);
    if (nbase0 + 32 < Nreal) epilogue(acc2, nbase0 + 32 + lh, mbase, lg, bias, done, prev, out, ldo, nt_store);
    if (nbase0 + 48 < Nreal) epilogue(acc3, nbase0 + 48 + lh, mbase, lg, bias, done, prev, out, ldo, nt_store);
}

// ---------------- GRU gate combine (biases already folded into gi/gh) ----------------
__global__ void k_gru(const float* __restrict__ gi, const float* __restrict__ gh,
                      float* __restrict__ h, _Float16* __restrict__ h16,
                      const int* __restrict__ done) {
    int t = blockIdx.x;
    int j = blockIdx.y * 256 + threadIdx.x;
    size_t b = (size_t)t * 3 * H_;
    float ir = gi[b + j], iz = gi[b + H_ + j], inn = gi[b + 2 * H_ + j];
    float hr = gh[b + j], hz = gh[b + H_ + j], hnn = gh[b + 2 * H_ + j];
    float r = 1.f / (1.f + __expf(-(ir + hr)));
    float z = 1.f / (1.f + __expf(-(iz + hz)));
    float n = tanhf(inn + r * hnn);
    float hold = h[(size_t)t * H_ + j];
    float hnew = (1.f - z) * n + z * hold;
    if (done && done[t]) hnew = hold;
    h[(size_t)t * H_ + j]   = hnew;
    h16[(size_t)t * H_ + j] = (_Float16)hnew;
}

// ---------------- per-token argmax over V (non-teacher-forcing fallback) ----------------
__global__ void k_argmax(const float* __restrict__ scores,  // base = d_out + step*V
                         int* __restrict__ amax) {
    __shared__ float sv[256];
    __shared__ int   si[256];
    int t = blockIdx.x, tid = threadIdx.x;
    const float* row = scores + (size_t)t * (size_t)F_ * V_;
    float best = -3.4e38f;
    int bi = 0;
    for (int v = tid; v < V_; v += 256) {
        float x = row[v];
        if (x > best) { best = x; bi = v; }
    }
    sv[tid] = best; si[tid] = bi;
    __syncthreads();
    for (int s = 128; s > 0; s >>= 1) {
        if (tid < s) {
            bool take = (sv[tid + s] > sv[tid]) ||
                        (sv[tid + s] == sv[tid] && si[tid + s] < si[tid]);
            if (take) { sv[tid] = sv[tid + s]; si[tid] = si[tid + s]; }
        }
        __syncthreads();
    }
    if (tid == 0) amax[t] = si[0];
}

// ---------------- advance carry: c, done (reads use_tf flag on-device; capture-safe) -----
__global__ void k_update(const int* __restrict__ out_chars, const int* __restrict__ amax,
                         const int* __restrict__ use_tf, int* __restrict__ c,
                         int* __restrict__ done, int step) {
    int t = threadIdx.x;
    int cn = (use_tf[0] != 0) ? out_chars[1 + t * F_ + step] : amax[t];
    if (cn == EOS_) done[t] = 1;
    c[t] = cn;
}

extern "C" void kernel_launch(void* const* d_in, const int* in_sizes, int n_in,
                              void* d_out, int out_size, void* d_ws, size_t ws_size,
                              hipStream_t stream) {
    (void)in_sizes; (void)n_in; (void)out_size; (void)ws_size;

    const float* token_ctx = (const float*)d_in[0];
    const float* char_emb  = (const float*)d_in[1];
    const float* encWih    = (const float*)d_in[2];
    const float* encWhh    = (const float*)d_in[3];
    const float* enc_bih   = (const float*)d_in[4];
    const float* enc_bhh   = (const float*)d_in[5];
    const float* decWih    = (const float*)d_in[6];
    const float* decWhh    = (const float*)d_in[7];
    const float* dec_bih   = (const float*)d_in[8];
    const float* dec_bhh   = (const float*)d_in[9];
    const float* outW      = (const float*)d_in[10];
    const float* out_b     = (const float*)d_in[11];
    const int*   sent      = (const int*)d_in[12];
    const int*   out_chars = (const int*)d_in[13];
    const int*   use_tf    = (const int*)d_in[16];
    float* out = (float*)d_out;

    // -------- workspace carve (256B aligned) --------
    char* p = (char*)d_ws;
    auto carve = [&](size_t bytes) -> char* {
        char* r = p;
        p += (bytes + 255) & ~(size_t)255;
        return r;
    };
    _Float16* emb16    = (_Float16*)carve((size_t)V_ * E_ * 2);        // 5.1 MB
    _Float16* encWih16 = (_Float16*)carve((size_t)3 * H_ * E_ * 2);    // 1.6 MB
    _Float16* encWhh16 = (_Float16*)carve((size_t)3 * H_ * H_ * 2);    // 6.3 MB
    _Float16* decWih16 = (_Float16*)carve((size_t)3 * H_ * E_ * 2);    // 1.6 MB
    _Float16* decWhh16 = (_Float16*)carve((size_t)3 * H_ * H_ * 2);    // 6.3 MB
    _Float16* outW16   = (_Float16*)carve((size_t)VPAD_ * H_ * 2);     // 20.6 MB (48 pad rows)
    float*    h        = (float*)carve((size_t)T_ * H_ * 4);           // 1 MB
    _Float16* h16      = (_Float16*)carve((size_t)T_ * H_ * 2);        // 0.5 MB
    _Float16* x16      = (_Float16*)carve((size_t)T_ * E_ * 2);        // 128 KB
    float*    gi       = (float*)carve((size_t)T_ * 3 * H_ * 4);       // 3 MB
    float*    gh       = (float*)carve((size_t)T_ * 3 * H_ * 4);       // 3 MB
    int*      c_cur    = (int*)carve((size_t)T_ * 4);
    int*      done     = (int*)carve((size_t)T_ * 4);
    int*      amax     = (int*)carve((size_t)T_ * 4);

    auto cvt = [&](const float* src, _Float16* dst, size_t n) {
        k_f32_to_f16<<<dim3((unsigned)((n + 255) / 256)), dim3(256), 0, stream>>>(src, dst, (int)n);
    };
    cvt(char_emb, emb16,    (size_t)V_ * E_);
    cvt(encWih,   encWih16, (size_t)3 * H_ * E_);
    cvt(encWhh,   encWhh16, (size_t)3 * H_ * H_);
    cvt(decWih,   decWih16, (size_t)3 * H_ * E_);
    cvt(decWhh,   decWhh16, (size_t)3 * H_ * H_);
    cvt(outW,     outW16,   (size_t)V_ * H_);   // pad rows never affect stored results

    k_h0<<<dim3(T_, H_ / 256), dim3(256), 0, stream>>>(token_ctx, sent, h, h16);
    k_c0<<<dim3(1), dim3(T_), 0, stream>>>(out_chars, c_cur, done);

    const dim3 wave(32);
    const dim3 gridGru(3 * H_ / 64, T_ / 16);    // 48 x 16 blocks (16x64 tiles)
    const dim3 gridOut(VPAD_ / 64, T_ / 16);     // 157 x 16 blocks
    const unsigned long long ld3h = 3ull * H_;
    const unsigned long long ldv  = (unsigned long long)F_ * V_;

    // ---------------- encoder: 8 GRU steps ----------------
    for (int s = 0; s < C_; ++s) {
        k_embed_enc<<<dim3(T_), dim3(E_), 0, stream>>>(sent, emb16, x16, s);
        k_wmma_gemm<<<gridGru, wave, 0, stream>>>(x16, encWih16, enc_bih,
                                                  nullptr, nullptr, gi, E_, 3 * H_, ld3h, 0);
        k_wmma_gemm<<<gridGru, wave, 0, stream>>>(h16, encWhh16, enc_bhh,
                                                  nullptr, nullptr, gh, H_, 3 * H_, ld3h, 0);
        k_gru<<<dim3(T_, H_ / 256), dim3(256), 0, stream>>>(gi, gh, h, h16, nullptr);
    }

    // ---------------- decoder: 32 steps ----------------
    for (int s = 0; s < F_; ++s) {
        k_embed_dec<<<dim3(T_), dim3(E_), 0, stream>>>(c_cur, emb16, x16);
        k_wmma_gemm<<<gridGru, wave, 0, stream>>>(x16, decWih16, dec_bih,
                                                  nullptr, nullptr, gi, E_, 3 * H_, ld3h, 0);
        k_wmma_gemm<<<gridGru, wave, 0, stream>>>(h16, decWhh16, dec_bhh,
                                                  nullptr, nullptr, gh, H_, 3 * H_, ld3h, 0);
        k_gru<<<dim3(T_, H_ / 256), dim3(256), 0, stream>>>(gi, gh, h, h16, done);
        // scores: out[t, s, :] = done ? out[t, s-1, :] (or 0) : h @ outW^T + b  (NT stores
        // keep the 10 MB/step score stream from evicting the L2-resident weight panel)
        const float* prev = (s > 0) ? (out + (size_t)(s - 1) * V_) : nullptr;
        k_wmma_gemm<<<gridOut, wave, 0, stream>>>(h16, outW16, out_b,
                                                  done, prev, out + (size_t)s * V_,
                                                  H_, V_, ldv, 1);
        k_argmax<<<dim3(T_), dim3(256), 0, stream>>>(out + (size_t)s * V_, amax);
        k_update<<<dim3(1), dim3(T_), 0, stream>>>(out_chars, amax, use_tf, c_cur, done, s);
    }
}
